// LSTMModel_30107720745843
// MI455X (gfx1250) — matrix operand order repaired and checked by reading, not compiled, not run on hardware
//
#include <hip/hip_runtime.h>

typedef __attribute__((ext_vector_type(16))) _Float16 v16h;
typedef __attribute__((ext_vector_type(8)))  _Float16 v8h;
typedef __attribute__((ext_vector_type(8)))  float    v8f;

#define B_   64
#define T_   2048
#define I_   32
#define H_   256
#define G_   1024   /* 4*H */
#define O_   32
#define NT_  64     /* 16-wide N tiles over G */
#define KT_  8      /* 32-wide K tiles over H */

/* workspace layout (bytes) */
#define OFF_WIHP  ((size_t)0)            /*  64 tiles * 512 f16 =   64 KB */
#define OFF_WHHP  ((size_t)65536)        /* 512 tiles * 512 f16 =  512 KB */
#define OFF_FCWP  ((size_t)589824)       /*  16 tiles * 512 f16 =   16 KB */
#define OFF_XP    ((size_t)606208)       /* 8192 tiles* 512 f16 =    8 MB */
#define OFF_XG    ((size_t)8994816)      /* B*T*4H f16          =  256 MB */
#define OFF_HH    ((size_t)277430272)    /* B*T*H  f16          =   64 MB */

/* scan kernel dynamic LDS: W_hh kt4..7 (256 KB) + h buffer (16x264 f16) */
#define LDS_WHH_BYTES  262144
#define LDS_TOTAL      (LDS_WHH_BYTES + 16 * 264 * 2)

#if __has_builtin(__builtin_amdgcn_tanhf)
#define FAST_TANH(x) __builtin_amdgcn_tanhf(x)
#elif __has_builtin(__builtin_amdgcn_tanh_f32)
#define FAST_TANH(x) __builtin_amdgcn_tanh_f32(x)
#else
#define FAST_TANH(x) tanhf(x)
#endif

__device__ __forceinline__ v8f wmma16(v16h a, v16h b, v8f c) {
    return __builtin_amdgcn_wmma_f32_16x16x32_f16(false, a, false, b,
                                                  (short)0, c, false, false);
}
__device__ __forceinline__ float sigm(float x) { return 1.0f / (1.0f + __expf(-x)); }

union V16U { v16h v; v8h h[2]; };

/* ---- pack W[rows x K] into WMMA B-matrix lane layout, B(k,n)=W[n,k] ----
   tile = nt*ktiles + kt; n=nt*16+(lane&15), h=lane/16;
   k=kt*32+8h+e (e<8), k=kt*32+16+8h+(e-8) (e>=8)                          */
__global__ void pack_bmat(const float* __restrict__ W, int K,
                          int ntiles, int ktiles, _Float16* __restrict__ out) {
    int w    = threadIdx.x >> 5;
    int lane = threadIdx.x & 31;
    int tile = blockIdx.x * 8 + w;
    if (tile >= ntiles * ktiles) return;
    int nt = tile / ktiles, kt = tile % ktiles;
    int n  = nt * 16 + (lane & 15);
    int kb = kt * 32 + ((lane < 16) ? 0 : 8);
    _Float16* o = out + (size_t)tile * 512 + lane * 16;
    const float* src = W + (size_t)n * K + kb;
    #pragma unroll
    for (int e = 0; e < 16; ++e) o[e] = (_Float16)src[e + ((e < 8) ? 0 : 8)];
}

/* ---- pack x[b,t,i] (f32) into WMMA A-matrix f16 tiles (16x32 per tile) -- */
__global__ void pack_x(const float* __restrict__ x, _Float16* __restrict__ xP) {
    int w    = threadIdx.x >> 5;
    int lane = threadIdx.x & 31;
    int tile = blockIdx.x * 8 + w;            /* = bblk*T + t, 8192 total */
    int bblk = tile >> 11;
    int t    = tile & 2047;
    int b    = bblk * 16 + (lane & 15);
    const float* src = x + ((size_t)b * T_ + t) * I_;
    _Float16* o = xP + (size_t)tile * 512 + lane * 16;
    int base = (lane < 16) ? 0 : 8;
    #pragma unroll
    for (int e = 0; e < 16; ++e) {
        int k = base + e + ((e < 8) ? 0 : 8);
        o[e] = (_Float16)src[k];
    }
}

/* ---- xg = x @ W_ih^T + (b_ih + b_hh), stored f16 in C-tile lane layout:
   xgP[((mtile*64)+nt)*256 + lane*8 + r] = xg[row=r+half][col]             */
__global__ void __launch_bounds__(256)
xg_gemm(const _Float16* __restrict__ xP, const _Float16* __restrict__ WihP,
        const float* __restrict__ b_ih, const float* __restrict__ b_hh,
        _Float16* __restrict__ xgP) {
    int w     = threadIdx.x >> 5;
    int lane  = threadIdx.x & 31;
    int mtile = blockIdx.x * 8 + w;           /* 0..8191 = bblk*T + t */
    int col   = lane & 15;
    v16h a = *(const v16h*)(xP + (size_t)mtile * 512 + lane * 16);
    for (int nt = 0; nt < NT_; ++nt) {
        v8f acc = {};
        v16h b = *(const v16h*)(WihP + (size_t)nt * 512 + lane * 16);
        acc = wmma16(a, b, acc);
        float bb = b_ih[nt * 16 + col] + b_hh[nt * 16 + col];
        v8h o;
        #pragma unroll
        for (int r = 0; r < 8; ++r) o[r] = (_Float16)(acc[r] + bb);
        *(v8h*)(xgP + ((size_t)mtile * NT_ + nt) * 256 + lane * 8) = o;
    }
}

/* ---- persistent recurrent scan: 4 workgroups x 512 threads (16 waves).
   Wave w owns hidden cols [16w,16w+16) and gate tiles {w,16+w,32+w,48+w}.
   W_hh kt0..3 resident in VGPRs (16 tiles = 128 regs), kt4..7 resident in
   LDS (256 KB); xg streams from global; c never leaves registers.         */
__global__ void __launch_bounds__(512, 1)
lstm_scan(const _Float16* __restrict__ xgP, const _Float16* __restrict__ WhhP,
          _Float16* __restrict__ hHist) {
    extern __shared__ char smem[];
    _Float16* WhhL = (_Float16*)smem;                           /* 256 KB */
    _Float16 (*hb)[264] = (_Float16 (*)[264])(smem + LDS_WHH_BYTES);

    const int w    = threadIdx.x >> 5;
    const int lane = threadIdx.x & 31;
    const int bblk = blockIdx.x;
    const int col  = lane & 15;
    const int rowA = lane & 15;
    const int half = (lane < 16) ? 0 : 8;

    for (int i = threadIdx.x; i < 16 * 264; i += 512)
        (&hb[0][0])[i] = (_Float16)0.0f;

    /* preload W_hh: kt0..3 -> registers, kt4..7 -> LDS */
    v16h breg[4][4];
    #pragma unroll
    for (int q = 0; q < 4; ++q) {
        int nt = q * 16 + w;
        #pragma unroll
        for (int kt = 0; kt < 4; ++kt)
            breg[q][kt] = *(const v16h*)(WhhP +
                          ((size_t)nt * KT_ + kt) * 512 + lane * 16);
        #pragma unroll
        for (int kt = 4; kt < 8; ++kt) {
            V16U u;
            u.v = *(const v16h*)(WhhP + ((size_t)nt * KT_ + kt) * 512 + lane * 16);
            size_t tb = ((size_t)((w * 4 + q) * 4 + (kt - 4))) * 512 + lane * 16;
            *(v8h*)(WhhL + tb)     = u.h[0];
            *(v8h*)(WhhL + tb + 8) = u.h[1];
        }
    }
    __syncthreads();

    v8f c = {};
    const size_t mbase = (size_t)bblk * T_;

    for (int t = 0; t < T_; ++t) {
        /* keep LDS-resident B tiles streamed, never re-hoisted to regs */
        asm volatile("" ::: "memory");

        /* accumulators start from precomputed xg (f16, C-tile layout) */
        v8f acc[4];
        #pragma unroll
        for (int q = 0; q < 4; ++q) {
            const _Float16* p = xgP + ((mbase + t) * NT_ + (q * 16 + w)) * 256
                                + lane * 8;
            v8h z = *(const v8h*)p;
            #pragma unroll
            for (int r = 0; r < 8; ++r) acc[q][r] = (float)z[r];
        }
        if (t + 1 < T_)
            __builtin_prefetch(xgP + ((mbase + t + 1) * NT_ + (w * 4)) * 256, 0, 0);

        /* h @ W_hh^T : kt-outer so only one A tile is live at a time */
        #pragma unroll
        for (int kt = 0; kt < KT_; ++kt) {
            int c0 = kt * 32 + half;
            V16U u;
            u.h[0] = *(const v8h*)&hb[rowA][c0];
            u.h[1] = *(const v8h*)&hb[rowA][c0 + 16];
            v16h a = u.v;
            if (kt < 4) {
                #pragma unroll
                for (int q = 0; q < 4; ++q)
                    acc[q] = wmma16(a, breg[q][kt], acc[q]);
            } else {
                #pragma unroll
                for (int q = 0; q < 4; ++q) {
                    size_t tb = ((size_t)((w * 4 + q) * 4 + (kt - 4))) * 512
                                + lane * 16;
                    V16U b;
                    b.h[0] = *(const v8h*)(WhhL + tb);
                    b.h[1] = *(const v8h*)(WhhL + tb + 8);
                    acc[q] = wmma16(a, b.v, acc[q]);
                }
            }
        }
        __syncthreads();                    /* all h_old reads complete */

        /* gates + cell update: i/f/g/o lane-aligned, pure VALU */
        v8f hv;
        #pragma unroll
        for (int r = 0; r < 8; ++r) {
            float iv = sigm(acc[0][r]);
            float fv = sigm(acc[1][r]);
            float gv = FAST_TANH(acc[2][r]);
            float ov = sigm(acc[3][r]);
            float cn = fv * c[r] + iv * gv;
            c[r]  = cn;
            hv[r] = ov * FAST_TANH(cn);
        }

        int hcol = w * 16 + col;
        #pragma unroll
        for (int r = 0; r < 8; ++r)
            hb[r + half][hcol] = (_Float16)hv[r];
        __syncthreads();                    /* h_new visible to all     */

        /* coalesced LDS -> global h history */
        {
            int row = threadIdx.x >> 5;
            int cc  = (threadIdx.x & 31) * 8;
            *(v8h*)&hHist[(((size_t)(bblk * 16 + row)) * T_ + t) * H_ + cc] =
                *(const v8h*)&hb[row][cc];
        }
    }
}

/* ---- output GEMM: out = hHist @ fc_w^T + fc_b ; M=B*T, N=32, K=256 ---- */
__global__ void __launch_bounds__(256)
out_gemm(const _Float16* __restrict__ hHist, const _Float16* __restrict__ fcwP,
         const float* __restrict__ fc_b, float* __restrict__ out) {
    const int w     = threadIdx.x >> 5;
    const int lane  = threadIdx.x & 31;
    const int mtile = blockIdx.x * 8 + w;      /* 0..8191 */
    const int rowA  = lane & 15;
    const int half  = (lane < 16) ? 0 : 8;
    const int col   = lane & 15;

    const size_t g = (size_t)mtile * 16 + rowA;
    v16h ah[KT_];
    #pragma unroll
    for (int kk = 0; kk < KT_; ++kk) {
        int c0 = kk * 32 + half;
        V16U u;
        u.h[0] = *(const v8h*)&hHist[g * H_ + c0];
        u.h[1] = *(const v8h*)&hHist[g * H_ + c0 + 16];
        ah[kk] = u.v;
    }
    #pragma unroll
    for (int nt = 0; nt < 2; ++nt) {
        v8f a = {};
        #pragma unroll
        for (int kk = 0; kk < KT_; ++kk) {
            v16h bh = *(const v16h*)(fcwP + ((size_t)nt * KT_ + kk) * 512
                                     + lane * 16);
            a = wmma16(ah[kk], bh, a);
        }
        float bb = fc_b[nt * 16 + col];
        #pragma unroll
        for (int r = 0; r < 8; ++r) {
            size_t m = (size_t)mtile * 16 + r + half;
            out[m * O_ + nt * 16 + col] = a[r] + bb;
        }
    }
}

extern "C" void kernel_launch(void* const* d_in, const int* in_sizes, int n_in,
                              void* d_out, int out_size, void* d_ws, size_t ws_size,
                              hipStream_t stream) {
    (void)in_sizes; (void)n_in; (void)out_size; (void)ws_size;
    const float* x    = (const float*)d_in[0];
    const float* W_ih = (const float*)d_in[1];
    const float* W_hh = (const float*)d_in[2];
    const float* b_ih = (const float*)d_in[3];
    const float* b_hh = (const float*)d_in[4];
    const float* fc_w = (const float*)d_in[5];
    const float* fc_b = (const float*)d_in[6];
    float* out = (float*)d_out;

    char* ws = (char*)d_ws;
    _Float16* WihP  = (_Float16*)(ws + OFF_WIHP);
    _Float16* WhhP  = (_Float16*)(ws + OFF_WHHP);
    _Float16* fcwP  = (_Float16*)(ws + OFF_FCWP);
    _Float16* xP    = (_Float16*)(ws + OFF_XP);
    _Float16* xgP   = (_Float16*)(ws + OFF_XG);
    _Float16* hHist = (_Float16*)(ws + OFF_HH);

    hipLaunchKernelGGL(pack_bmat, dim3(8),    dim3(256), 0, stream, W_ih, I_, NT_, 1,   WihP);
    hipLaunchKernelGGL(pack_bmat, dim3(64),   dim3(256), 0, stream, W_hh, H_, NT_, KT_, WhhP);
    hipLaunchKernelGGL(pack_bmat, dim3(2),    dim3(256), 0, stream, fc_w, H_, 2,   KT_, fcwP);
    hipLaunchKernelGGL(pack_x,    dim3(1024), dim3(256), 0, stream, x, xP);
    hipLaunchKernelGGL(xg_gemm,   dim3(1024), dim3(256), 0, stream,
                       xP, WihP, b_ih, b_hh, xgP);
    hipLaunchKernelGGL(lstm_scan, dim3(4),    dim3(512), LDS_TOTAL, stream,
                       xgP, WhhP, hHist);
    hipLaunchKernelGGL(out_gemm,  dim3(1024), dim3(256), 0, stream,
                       hHist, fcwP, fc_b, out);
}
